// JellyfishBrain_15195594293738
// MI455X (gfx1250) — matrix operand order repaired and checked by reading, not compiled, not run on hardware
//
#include <hip/hip_runtime.h>
#include <math.h>

typedef __attribute__((ext_vector_type(16))) __bf16 v16bf;
typedef __attribute__((ext_vector_type(8)))  __bf16 v8bf;
typedef __attribute__((ext_vector_type(8)))  float  v8f;

#define WMMA_BF16(a, b, c) \
  __builtin_amdgcn_wmma_f32_16x16x32_bf16(false, (a), false, (b), (short)0, (c), false, false)

// ---- model dimensions ----
#define Bm   128
#define Tm   512
#define DIN  32
#define Hm   512
#define NRH  6
#define RHm  61
#define NHm  81
#define MHm  61
// TOTAL = 6*61 + 81 + 61 = 508

// ---------------------------------------------------------------------------
// WMMA fragment helpers (bf16 16x16x32, wave32)
// A (16xK row-major in LDS): lane holds row lane&15; lanes<16 carry K
// {kb..kb+7, kb+16..kb+23}, lanes>=16 carry {kb+8..kb+15, kb+24..kb+31}.
// B packed per 32x16 block: lane holds 16 contiguous bf16 = col lane&15,
// K rows kb+(lane<16?0:8)+{0..7,16..23} (same k order as A).
// C: vgpr i -> row i + 8*(lane>=16), col lane&15.
// ---------------------------------------------------------------------------
__device__ __forceinline__ v16bf load_a_frag(const __bf16* A, int astride, int lane, int kbase) {
  const __bf16* p = A + (lane & 15) * astride + kbase + ((lane >> 4) & 1) * 8;
  v8bf lo = *(const v8bf*)p;
  v8bf hi = *(const v8bf*)(p + 16);
  v16bf a;
#pragma unroll
  for (int j = 0; j < 8; ++j) { a[j] = lo[j]; a[8 + j] = hi[j]; }
  return a;
}

__device__ __forceinline__ void store_c(float* C, int cstride, int lane, v8f acc) {
  int col = lane & 15, h8 = ((lane >> 4) & 1) * 8;
#pragma unroll
  for (int i = 0; i < 8; ++i) C[(size_t)(i + h8) * cstride + col] = acc[i];
}

// C(16 x NT*16) = A(16 x KT*32) @ Wpacked; waves split N-tiles.
template <int KT, int NT>
__device__ __forceinline__ void gemm_lds(const __bf16* A, int astride,
                                         const __bf16* __restrict__ Wp,
                                         float* C, int cstride,
                                         int lane, int wave, int nwaves) {
  for (int nt = wave; nt < NT; nt += nwaves) {
    v8f acc = {};
#pragma unroll
    for (int kt = 0; kt < KT; ++kt) {
      v16bf a = load_a_frag(A, astride, lane, kt * 32);
      v16bf b = *(const v16bf*)(Wp + ((size_t)kt * NT + nt) * 512 + lane * 16);
      acc = WMMA_BF16(a, b, acc);
    }
    store_c(C + nt * 16, cstride, lane, acc);
  }
}

__device__ __forceinline__ float sigmoidf_(float v) { return 1.f / (1.f + __expf(-v)); }
__device__ __forceinline__ float geluf_(float v) { return 0.5f * v * (1.f + erff(v * 0.70710678f)); }

// Row-wise LayerNorm in place on S[16][stride] (F valid features).
__device__ void ln_rows(float* S, int stride, int F, const float* g, const float* be,
                        float* red, int tid, int nthreads) {
  if (tid < 16) {
    const float* row = S + tid * stride;
    float s = 0.f, s2 = 0.f;
    for (int j = 0; j < F; ++j) { float v = row[j]; s += v; s2 += v * v; }
    float m = s / F;
    red[tid * 2 + 0] = m;
    red[tid * 2 + 1] = rsqrtf(s2 / F - m * m + 1e-5f);
  }
  __syncthreads();
  for (int idx = tid; idx < 16 * F; idx += nthreads) {
    int r = idx / F, j = idx - r * F;
    S[r * stride + j] = (S[r * stride + j] - red[r * 2]) * red[r * 2 + 1] * g[j] + be[j];
  }
  __syncthreads();
}

// ---------------------------------------------------------------------------
// Weight packing: f32 KxN row-major -> bf16 WMMA-B blocks [kt][nt] of 512.
// ---------------------------------------------------------------------------
__global__ void pack_kernel(const float* __restrict__ src, __bf16* __restrict__ dst,
                            int K, int N, int KT, int NT) {
  int idx = blockIdx.x * blockDim.x + threadIdx.x;
  int total = KT * NT * 512;
  if (idx >= total) return;
  int blk = idx >> 9;
  int e = idx & 511;
  int lane = e >> 4;
  int j = e & 15;
  int kt = blk / NT, nt = blk - kt * NT;
  int col = nt * 16 + (lane & 15);
  int row = kt * 32 + ((lane < 16) ? 0 : 8) + (j & 7) + ((j >> 3) << 4);
  float v = (row < K && col < N) ? src[(size_t)row * N + col] : 0.f;
  dst[idx] = (__bf16)v;
}

// ---------------------------------------------------------------------------
// Persistent RNN kernel: 8 workgroups, each owns a 16-row batch tile.
// ---------------------------------------------------------------------------
struct RnnArgs {
  const float* x;
  const float *in_b, *coup;
  const float *ax_pb, *ax_gb;
  const float *rh_b, *rh_ls, *rh_g, *rh_be;
  const float *net_b, *net_ls, *net_g, *net_be;
  const float *mot_b, *mot_ls, *mot_g, *mot_be;
  const float *th_b, *th_g, *th_be, *tp_b;
  const __bf16 *Pinw, *Paxp, *Paxg, *Prh, *Pnet, *Pmot, *Pth, *Ptp;
  float* hid;
};

__global__ __launch_bounds__(256) void rnn_kernel(RnnArgs A) {
  const int tid = threadIdx.x;
  const int lane = tid & 31;
  const int wave = tid >> 5;
  const int b0 = blockIdx.x * 16;

  __shared__ float  s_rh[NRH][16][64];     // ring states (padded cols stay 0)
  __shared__ float  s_net[16][96];
  __shared__ float  s_mot[16][64];
  __shared__ __bf16 inp_bf[16][64];
  __shared__ __bf16 ax_bf[NRH][16][64];
  __shared__ __bf16 Abf[16][544];          // staged A operand (max K = 544)
  __shared__ float  fbuf[16][512];         // GEMM C / pre-activation scratch
  __shared__ __bf16 gate_bf[16][512];
  __shared__ float  red[32];
  __shared__ float  coup_s[8];

  for (int i = tid; i < NRH * 16 * 64; i += 256) (&s_rh[0][0][0])[i] = 0.f;
  for (int i = tid; i < 16 * 96; i += 256) (&s_net[0][0])[i] = 0.f;
  for (int i = tid; i < 16 * 64; i += 256) (&s_mot[0][0])[i] = 0.f;
  if (tid < NRH) coup_s[tid] = sigmoidf_(A.coup[tid]);
  __syncthreads();

  for (int t = 0; t < Tm; ++t) {
    // ---- input projection: inp = x_t @ in_w + in_b ----
    for (int i = tid; i < 16 * DIN; i += 256) {
      int r = i >> 5, d = i & 31;
      Abf[r][d] = (__bf16)A.x[((size_t)(b0 + r) * Tm + t) * DIN + d];
    }
    __syncthreads();
    gemm_lds<1, 4>(&Abf[0][0], 544, A.Pinw, &fbuf[0][0], 512, lane, wave, 8);
    __syncthreads();
    for (int i = tid; i < 16 * 64; i += 256) {
      int r = i >> 6, c = i & 63;
      float v = (c < RHm) ? fbuf[r][c] + A.in_b[c] : 0.f;
      inp_bf[r][c] = (__bf16)v;
    }
    __syncthreads();

    // ---- axon gating per ring (from OLD s_rh) ----
    for (int n = 0; n < NRH; ++n) {
      for (int i = tid; i < 16 * 64; i += 256) {
        int r = i >> 6, c = i & 63;
        Abf[r][c] = (__bf16)s_rh[n][r][c];
      }
      __syncthreads();
      gemm_lds<2, 4>(&Abf[0][0], 544, A.Paxp + n * (2 * 4 * 512), &fbuf[0][0], 512, lane, wave, 8);
      gemm_lds<2, 4>(&Abf[0][0], 544, A.Paxg + n * (2 * 4 * 512), &fbuf[0][64], 512, lane, wave, 8);
      __syncthreads();
      for (int i = tid; i < 16 * 64; i += 256) {
        int r = i >> 6, c = i & 63;
        float av = 0.f;
        if (c < RHm) {
          float pv = tanhf(fbuf[r][c] + A.ax_pb[n * RHm + c]);
          float gv = sigmoidf_(fbuf[r][64 + c] + A.ax_gb[n * RHm + c]);
          av = gv * pv;
        }
        ax_bf[n][r][c] = (__bf16)av;
      }
      __syncthreads();
    }

    // ---- ring-cell update (cell_in = [inp, left, right, s, s]) ----
    for (int n = 0; n < NRH; ++n) {
      int nl = (n + NRH - 1) % NRH;
      int nr = (n + 1) % NRH;
      float cl = coup_s[nl], cr = coup_s[n];
      for (int i = tid; i < 16 * 320; i += 256) {
        int r = i / 320, c = i - r * 320;
        float v = 0.f;
        if      (c < 61)  v = (float)inp_bf[r][c];
        else if (c < 122) v = cl * (float)ax_bf[nl][r][c - 61];
        else if (c < 183) v = cr * (float)ax_bf[nr][r][c - 122];
        else if (c < 244) v = s_rh[n][r][c - 183];
        else if (c < 305) v = s_rh[n][r][c - 244];
        Abf[r][c] = (__bf16)v;
      }
      __syncthreads();
      gemm_lds<10, 4>(&Abf[0][0], 544, A.Prh + n * (10 * 4 * 512), &fbuf[0][0], 512, lane, wave, 8);
      __syncthreads();
      for (int i = tid; i < 16 * RHm; i += 256) {
        int r = i / RHm, c = i - r * RHm;
        float pre = tanhf(fbuf[r][c] + A.rh_b[n * RHm + c]);
        float s = s_rh[n][r][c];
        s_rh[n][r][c] = s + __expf(A.rh_ls[n * RHm + c]) * (pre - s);
      }
      __syncthreads();
      ln_rows(&s_rh[n][0][0], 64, RHm, A.rh_g + n * RHm, A.rh_be + n * RHm, red, tid, 256);
    }

    // ---- network layer: cin = [rh_flat, s_net, s_net] (K=528->544) ----
    for (int i = tid; i < 16 * 544; i += 256) {
      int r = i / 544, c = i - r * 544;
      float v = 0.f;
      if      (c < 366) { int n = c / 61, j = c - n * 61; v = s_rh[n][r][j]; }
      else if (c < 447) v = s_net[r][c - 366];
      else if (c < 528) v = s_net[r][c - 447];
      Abf[r][c] = (__bf16)v;
    }
    __syncthreads();
    gemm_lds<17, 6>(&Abf[0][0], 544, A.Pnet, &fbuf[0][0], 512, lane, wave, 8);
    __syncthreads();
    for (int i = tid; i < 16 * NHm; i += 256) {
      int r = i / NHm, c = i - r * NHm;
      float pre = tanhf(fbuf[r][c] + A.net_b[c]);
      float s = s_net[r][c];
      s_net[r][c] = s + 0.5f * __expf(A.net_ls[c]) * (pre - s);
    }
    __syncthreads();
    ln_rows(&s_net[0][0], 96, NHm, A.net_g, A.net_be, red, tid, 256);

    // ---- motor layer: cin = [s_net, s_mot] (K=142->160) ----
    for (int i = tid; i < 16 * 160; i += 256) {
      int r = i / 160, c = i - r * 160;
      float v = 0.f;
      if      (c < 81)  v = s_net[r][c];
      else if (c < 142) v = s_mot[r][c - 81];
      Abf[r][c] = (__bf16)v;
    }
    __syncthreads();
    gemm_lds<5, 4>(&Abf[0][0], 544, A.Pmot, &fbuf[0][0], 512, lane, wave, 8);
    __syncthreads();
    for (int i = tid; i < 16 * MHm; i += 256) {
      int r = i / MHm, c = i - r * MHm;
      float pre = tanhf(fbuf[r][c] + A.mot_b[c]);
      float s = s_mot[r][c];
      s_mot[r][c] = s + 1.2f * __expf(A.mot_ls[c]) * (pre - s);
    }
    __syncthreads();
    ln_rows(&s_mot[0][0], 64, MHm, A.mot_g, A.mot_be, red, tid, 256);

    // ---- gated output head: allr(508->512) @ th_w / tp_w ----
    for (int i = tid; i < 16 * 512; i += 256) {
      int r = i >> 9, c = i & 511;
      float v = 0.f;
      if      (c < 366) { int n = c / 61, j = c - n * 61; v = s_rh[n][r][j]; }
      else if (c < 447) v = s_net[r][c - 366];
      else if (c < 508) v = s_mot[r][c - 447];
      Abf[r][c] = (__bf16)v;
    }
    __syncthreads();
    gemm_lds<16, 32>(&Abf[0][0], 544, A.Pth, &fbuf[0][0], 512, lane, wave, 8);
    __syncthreads();
    for (int i = tid; i < 16 * 512; i += 256) {
      int r = i >> 9, c = i & 511;
      fbuf[r][c] += A.th_b[c];
    }
    __syncthreads();
    ln_rows(&fbuf[0][0], 512, 512, A.th_g, A.th_be, red, tid, 256);
    for (int i = tid; i < 16 * 512; i += 256) {
      int r = i >> 9, c = i & 511;
      gate_bf[r][c] = (__bf16)sigmoidf_(fbuf[r][c]);
    }
    __syncthreads();
    gemm_lds<16, 32>(&Abf[0][0], 544, A.Ptp, &fbuf[0][0], 512, lane, wave, 8);
    __syncthreads();
    for (int i = tid; i < 16 * 512; i += 256) {
      int r = i >> 9, c = i & 511;
      float v = (float)gate_bf[r][c] * (fbuf[r][c] + A.tp_b[c]);
      A.hid[((size_t)(b0 + r) * Tm + t) * Hm + c] = v;
    }
    __syncthreads();
  }
}

// ---------------------------------------------------------------------------
// anomaly = sigmoid(LN(hidden) @ an_w + an_b); wave-per-16-row tile.
// ---------------------------------------------------------------------------
__global__ __launch_bounds__(256) void anomaly_kernel(
    const float* __restrict__ hid, const float* __restrict__ an_g,
    const float* __restrict__ an_be, const __bf16* __restrict__ Pan,
    const float* __restrict__ an_b, float* __restrict__ out) {
  int lane = threadIdx.x & 31, wave = threadIdx.x >> 5;
  int tile = blockIdx.x * 8 + wave;              // 4096 tiles total
  int rr = lane & 15, h8 = ((lane >> 4) & 1) * 8;
  const float* hrow = hid + (size_t)(tile * 16 + rr) * Hm;
  float s = 0.f, s2 = 0.f;
  for (int k = 0; k < Hm; ++k) { float v = hrow[k]; s += v; s2 += v * v; }
  float m = s * (1.f / Hm);
  float inv = rsqrtf(s2 * (1.f / Hm) - m * m + 1e-5f);
  v8f acc = {};
  for (int kt = 0; kt < 16; ++kt) {
    v16bf a;
#pragma unroll
    for (int j = 0; j < 8; ++j) {
      int k0 = kt * 32 + h8 + j, k1 = k0 + 16;
      a[j]     = (__bf16)((hrow[k0] - m) * inv * an_g[k0] + an_be[k0]);
      a[8 + j] = (__bf16)((hrow[k1] - m) * inv * an_g[k1] + an_be[k1]);
    }
    v16bf b = *(const v16bf*)(Pan + (size_t)kt * 512 + lane * 16);
    acc = WMMA_BF16(a, b, acc);
  }
  if (rr == 0) {
    float ab = an_b[0];
#pragma unroll
    for (int i = 0; i < 8; ++i)
      out[tile * 16 + i + h8] = sigmoidf_(acc[i] + ab);
  }
}

// ---------------------------------------------------------------------------
// predictions = LN(hidden[:, :-1]) @ ns_w + ns_b ; surprise = mse vs x[:,1:]
// ---------------------------------------------------------------------------
__global__ __launch_bounds__(256) void pred_kernel(
    const float* __restrict__ hid, const float* __restrict__ x,
    const float* __restrict__ ns_g, const float* __restrict__ ns_be,
    const float* __restrict__ ns_b, const __bf16* __restrict__ Pns,
    float* __restrict__ pred, float* __restrict__ sur) {
  int lane = threadIdx.x & 31, wave = threadIdx.x >> 5;
  int tile = blockIdx.x * 8 + wave;              // 4088 tiles total
  int rr = lane & 15, h8 = ((lane >> 4) & 1) * 8;
  int prow = tile * 16 + rr;
  int bb = prow / 511, tt = prow - bb * 511;
  const float* hrow = hid + ((size_t)bb * Tm + tt) * Hm;
  float s = 0.f, s2 = 0.f;
  for (int k = 0; k < Hm; ++k) { float v = hrow[k]; s += v; s2 += v * v; }
  float m = s * (1.f / Hm);
  float inv = rsqrtf(s2 * (1.f / Hm) - m * m + 1e-5f);
  v8f acc0 = {}, acc1 = {};
  for (int kt = 0; kt < 16; ++kt) {
    v16bf a;
#pragma unroll
    for (int j = 0; j < 8; ++j) {
      int k0 = kt * 32 + h8 + j, k1 = k0 + 16;
      a[j]     = (__bf16)((hrow[k0] - m) * inv * ns_g[k0] + ns_be[k0]);
      a[8 + j] = (__bf16)((hrow[k1] - m) * inv * ns_g[k1] + ns_be[k1]);
    }
    v16bf b0 = *(const v16bf*)(Pns + ((size_t)kt * 2 + 0) * 512 + lane * 16);
    v16bf b1 = *(const v16bf*)(Pns + ((size_t)kt * 2 + 1) * 512 + lane * 16);
    acc0 = WMMA_BF16(a, b0, acc0);
    acc1 = WMMA_BF16(a, b1, acc1);
  }
  float nb0 = ns_b[rr], nb1 = ns_b[16 + rr];
  float err[8];
#pragma unroll
  for (int i = 0; i < 8; ++i) {
    int orow = tile * 16 + i + h8;
    int ob = orow / 511, ot = orow - ob * 511;
    float p0 = acc0[i] + nb0;
    float p1 = acc1[i] + nb1;
    pred[(size_t)orow * 32 + rr] = p0;
    pred[(size_t)orow * 32 + 16 + rr] = p1;
    float x0 = x[((size_t)ob * Tm + ot + 1) * DIN + rr];
    float x1 = x[((size_t)ob * Tm + ot + 1) * DIN + 16 + rr];
    float d0 = p0 - x0, d1 = p1 - x1;
    err[i] = d0 * d0 + d1 * d1;
  }
#pragma unroll
  for (int mk = 1; mk <= 8; mk <<= 1)
#pragma unroll
    for (int i = 0; i < 8; ++i) err[i] += __shfl_xor(err[i], mk, 32);
  if (rr == 0) {
#pragma unroll
    for (int i = 0; i < 8; ++i)
      sur[tile * 16 + i + h8] = err[i] * (1.f / 32.f);
  }
}

// ---------------------------------------------------------------------------
// properties = gelu(LN(hidden[:, -1]) @ pr_w1 + b1) @ pr_w2 + b2
// ---------------------------------------------------------------------------
__global__ __launch_bounds__(256) void prop_kernel(
    const float* __restrict__ hid,
    const float* __restrict__ pr_g, const float* __restrict__ pr_be,
    const float* __restrict__ pr_b1, const float* __restrict__ pr_b2,
    const __bf16* __restrict__ Pp1, const __bf16* __restrict__ Pp2,
    float* __restrict__ props) {
  __shared__ __bf16 ph[8][16][256];
  int lane = threadIdx.x & 31, wave = threadIdx.x >> 5;
  int rr = lane & 15, h8 = ((lane >> 4) & 1) * 8;
  int bb = wave * 16 + rr;
  const float* hrow = hid + ((size_t)bb * Tm + (Tm - 1)) * Hm;
  float s = 0.f, s2 = 0.f;
  for (int k = 0; k < Hm; ++k) { float v = hrow[k]; s += v; s2 += v * v; }
  float m = s * (1.f / Hm);
  float inv = rsqrtf(s2 * (1.f / Hm) - m * m + 1e-5f);
  for (int nt = 0; nt < 16; ++nt) {
    v8f acc = {};
    for (int kt = 0; kt < 16; ++kt) {
      v16bf a;
#pragma unroll
      for (int j = 0; j < 8; ++j) {
        int k0 = kt * 32 + h8 + j, k1 = k0 + 16;
        a[j]     = (__bf16)((hrow[k0] - m) * inv * pr_g[k0] + pr_be[k0]);
        a[8 + j] = (__bf16)((hrow[k1] - m) * inv * pr_g[k1] + pr_be[k1]);
      }
      v16bf b = *(const v16bf*)(Pp1 + ((size_t)kt * 16 + nt) * 512 + lane * 16);
      acc = WMMA_BF16(a, b, acc);
    }
#pragma unroll
    for (int i = 0; i < 8; ++i)
      ph[wave][i + h8][nt * 16 + rr] = (__bf16)(acc[i] + pr_b1[nt * 16 + rr]);
  }
  __syncthreads();
  v8f acc2 = {};
  for (int kt = 0; kt < 8; ++kt) {
    v16bf a;
#pragma unroll
    for (int j = 0; j < 8; ++j) {
      int k0 = kt * 32 + h8 + j, k1 = k0 + 16;
      a[j]     = (__bf16)geluf_((float)ph[wave][rr][k0]);
      a[8 + j] = (__bf16)geluf_((float)ph[wave][rr][k1]);
    }
    v16bf b = *(const v16bf*)(Pp2 + (size_t)kt * 512 + lane * 16);
    acc2 = WMMA_BF16(a, b, acc2);
  }
  if (rr < 2) {
#pragma unroll
    for (int i = 0; i < 8; ++i)
      props[(size_t)(wave * 16 + i + h8) * 2 + rr] = acc2[i] + pr_b2[rr];
  }
}

// ---------------------------------------------------------------------------
// Host launcher
// ---------------------------------------------------------------------------
extern "C" void kernel_launch(void* const* d_in, const int* in_sizes, int n_in,
                              void* d_out, int out_size, void* d_ws, size_t ws_size,
                              hipStream_t stream) {
  (void)in_sizes; (void)n_in; (void)out_size; (void)ws_size;
  const float* x      = (const float*)d_in[0];
  const float* in_w   = (const float*)d_in[1];
  const float* in_b   = (const float*)d_in[2];
  const float* coup   = (const float*)d_in[3];
  const float* ax_pw  = (const float*)d_in[4];
  const float* ax_pb  = (const float*)d_in[5];
  const float* ax_gw  = (const float*)d_in[6];
  const float* ax_gb  = (const float*)d_in[7];
  const float* rh_w   = (const float*)d_in[8];
  const float* rh_b   = (const float*)d_in[9];
  const float* rh_ls  = (const float*)d_in[10];
  const float* rh_g   = (const float*)d_in[11];
  const float* rh_be  = (const float*)d_in[12];
  const float* net_w  = (const float*)d_in[13];
  const float* net_b  = (const float*)d_in[14];
  const float* net_ls = (const float*)d_in[15];
  const float* net_g  = (const float*)d_in[16];
  const float* net_be = (const float*)d_in[17];
  const float* mot_w  = (const float*)d_in[18];
  const float* mot_b  = (const float*)d_in[19];
  const float* mot_ls = (const float*)d_in[20];
  const float* mot_g  = (const float*)d_in[21];
  const float* mot_be = (const float*)d_in[22];
  const float* th_w   = (const float*)d_in[23];
  const float* th_b   = (const float*)d_in[24];
  const float* th_g   = (const float*)d_in[25];
  const float* th_be  = (const float*)d_in[26];
  const float* tp_w   = (const float*)d_in[27];
  const float* tp_b   = (const float*)d_in[28];
  const float* ns_g   = (const float*)d_in[29];
  const float* ns_be  = (const float*)d_in[30];
  const float* ns_w   = (const float*)d_in[31];
  const float* ns_b   = (const float*)d_in[32];
  const float* an_g   = (const float*)d_in[33];
  const float* an_be  = (const float*)d_in[34];
  const float* an_w   = (const float*)d_in[35];
  const float* an_b   = (const float*)d_in[36];
  const float* pr_g   = (const float*)d_in[37];
  const float* pr_be  = (const float*)d_in[38];
  const float* pr_w1  = (const float*)d_in[39];
  const float* pr_b1  = (const float*)d_in[40];
  const float* pr_w2  = (const float*)d_in[41];
  const float* pr_b2  = (const float*)d_in[42];

  // packed weight layout in workspace (bf16 elements)
  __bf16* ws = (__bf16*)d_ws;
  __bf16* Pinw = ws + 0;        // 1*4*512
  __bf16* Paxp = ws + 2048;     // 6 * 2*4*512
  __bf16* Paxg = ws + 26624;
  __bf16* Prh  = ws + 51200;    // 6 * 10*4*512
  __bf16* Pnet = ws + 174080;   // 17*6*512
  __bf16* Pmot = ws + 226304;   // 5*4*512
  __bf16* Pth  = ws + 236544;   // 16*32*512
  __bf16* Ptp  = ws + 498688;
  __bf16* Pns  = ws + 760832;   // 16*2*512
  __bf16* Pan  = ws + 777216;   // 16*1*512
  __bf16* Pp1  = ws + 785408;   // 16*16*512
  __bf16* Pp2  = ws + 916480;   // 8*1*512

  auto pack = [&](const float* src, __bf16* dst, int K, int N, int KT, int NT) {
    int total = KT * NT * 512;
    pack_kernel<<<(total + 255) / 256, 256, 0, stream>>>(src, dst, K, N, KT, NT);
  };
  pack(in_w, Pinw, 32, 61, 1, 4);
  for (int n = 0; n < 6; ++n) {
    pack(ax_pw + n * 61 * 61, Paxp + n * 4096, 61, 61, 2, 4);
    pack(ax_gw + n * 61 * 61, Paxg + n * 4096, 61, 61, 2, 4);
    pack(rh_w + n * 305 * 61, Prh + n * 20480, 305, 61, 10, 4);
  }
  pack(net_w, Pnet, 528, 81, 17, 6);
  pack(mot_w, Pmot, 142, 61, 5, 4);
  pack(th_w, Pth, 508, 512, 16, 32);
  pack(tp_w, Ptp, 508, 512, 16, 32);
  pack(ns_w, Pns, 512, 32, 16, 2);
  pack(an_w, Pan, 512, 1, 16, 1);
  pack(pr_w1, Pp1, 512, 256, 16, 16);
  pack(pr_w2, Pp2, 256, 2, 8, 1);

  // output slices (floats): pred | hidden | anomaly | surprise | properties
  float* outf  = (float*)d_out;
  float* pred  = outf;
  float* hid   = outf + 2093056;   // 128*511*32
  float* ano   = outf + 35647488;  // + 128*512*512
  float* sur   = outf + 35713024;  // + 128*512
  float* props = outf + 35778432;  // + 128*511

  RnnArgs ra;
  ra.x = x; ra.in_b = in_b; ra.coup = coup;
  ra.ax_pb = ax_pb; ra.ax_gb = ax_gb;
  ra.rh_b = rh_b; ra.rh_ls = rh_ls; ra.rh_g = rh_g; ra.rh_be = rh_be;
  ra.net_b = net_b; ra.net_ls = net_ls; ra.net_g = net_g; ra.net_be = net_be;
  ra.mot_b = mot_b; ra.mot_ls = mot_ls; ra.mot_g = mot_g; ra.mot_be = mot_be;
  ra.th_b = th_b; ra.th_g = th_g; ra.th_be = th_be; ra.tp_b = tp_b;
  ra.Pinw = Pinw; ra.Paxp = Paxp; ra.Paxg = Paxg; ra.Prh = Prh;
  ra.Pnet = Pnet; ra.Pmot = Pmot; ra.Pth = Pth; ra.Ptp = Ptp;
  ra.hid = hid;

  rnn_kernel<<<8, 256, 0, stream>>>(ra);
  anomaly_kernel<<<512, 256, 0, stream>>>(hid, an_g, an_be, Pan, an_b, ano);
  pred_kernel<<<511, 256, 0, stream>>>(hid, x, ns_g, ns_be, ns_b, Pns, pred, sur);
  prop_kernel<<<1, 256, 0, stream>>>(hid, pr_g, pr_be, pr_b1, pr_b2, Pp1, Pp2, props);
}
